// ODE_RNN_44616120271294
// MI455X (gfx1250) — matrix operand order repaired and checked by reading, not compiled, not run on hardware
//
#include <hip/hip_runtime.h>

// Problem constants (from reference)
#define LT   128   // sequence length
#define NB   1024  // batch
#define HH   256   // H
#define DD   512   // D = 2H
#define GG   1536  // 3D
#define OH   128   // ODE hidden
#define INU  32    // input units (ext has IN_U+1 = 33 cols, last is dt)
#define NOUT 16

typedef unsigned short u16t;
typedef __attribute__((ext_vector_type(16))) __bf16 v16bf;
typedef __attribute__((ext_vector_type(16))) u16t  v16u;
typedef __attribute__((ext_vector_type(8)))  u16t  v8u;
typedef __attribute__((ext_vector_type(8)))  float v8f;

union FragU { v16u u; v16bf b; v8u h[2]; };

// Native f32 -> bf16 (hardware v_cvt_*_bf16_f32 on gfx1250; RTNE)
__device__ __forceinline__ u16t f2bf(float x) {
  __bf16 b = (__bf16)x;
  return __builtin_bit_cast(u16t, b);
}
__device__ __forceinline__ float sigm(float x) { return 1.0f / (1.0f + __expf(-x)); }
__device__ __forceinline__ float tanh_f(float x) {
  x = fminf(fmaxf(x, -15.0f), 15.0f);
  float e = __expf(2.0f * x);
  return (e - 1.0f) / (e + 1.0f);
}

// A-fragment (16x32, bf16): lane holds row (lane&15); lanes 0-15 carry K={k0..k0+7, k0+16..k0+23},
// lanes 16-31 carry K={k0+8..k0+15, k0+24..k0+31} -> two contiguous 16B loads per lane.
__device__ __forceinline__ v16u lda(const u16t* base, int row, int stride, int k0, int lane) {
  const u16t* p = base + (size_t)row * stride + k0 + ((lane >> 4) << 3);
  FragU f;
  f.h[0] = *reinterpret_cast<const v8u*>(p);
  f.h[1] = *reinterpret_cast<const v8u*>(p + 16);
  return f.u;
}
// B-fragment (32x16, bf16): lane holds col (lane&15); lanes 0-15 K={k0..k0+7, k0+16..k0+23},
// lanes 16-31 K={k0+8..k0+15, k0+24..k0+31} -> two contiguous 16B loads per lane (weights stored N x K row-major).
__device__ __forceinline__ v16u ldb(const u16t* base, int nrow, int stride, int k0, int lane) {
  const u16t* p = base + (size_t)nrow * stride + k0 + ((lane >> 4) << 3);
  FragU f;
  f.h[0] = *reinterpret_cast<const v8u*>(p);
  f.h[1] = *reinterpret_cast<const v8u*>(p + 16);
  return f.u;
}
__device__ __forceinline__ v8f wmma_bf(v16u a, v16u b, v8f c) {
  FragU A, B; A.u = a; B.u = b;
  return __builtin_amdgcn_wmma_f32_16x16x32_bf16(false, A.b, false, B.b, (short)0, c, false, false);
}

// ---------------- one-time: weight convert / transpose to bf16 N x K ----------------
__global__ void k_convert(const float* Wih, const float* Whh, const float* Wu, const float* Wx,
                          const float* W1, const float* W2, const float* WLy,
                          u16t* wih, u16t* whh, u16t* wut, u16t* wxt,
                          u16t* w1t, u16t* w2t, u16t* wlyt) {
  int tid = blockIdx.x * blockDim.x + threadIdx.x;
  int np = gridDim.x * blockDim.x;
  for (int i = tid; i < GG * DD; i += np) { wih[i] = f2bf(Wih[i]); whh[i] = f2bf(Whh[i]); }
  for (int i = tid; i < HH * INU; i += np) { int n = i / INU, k = i % INU; wut[i] = f2bf(Wu[k * HH + n]); }
  for (int i = tid; i < HH * 32; i += np) { int n = i / 32, k = i % 32; wxt[i] = (k < NOUT) ? f2bf(Wx[k * HH + n]) : (u16t)0; }
  for (int i = tid; i < OH * HH; i += np) { int n = i / HH, k = i % HH; w1t[i] = f2bf(W1[k * OH + n]); }
  for (int i = tid; i < HH * OH; i += np) { int n = i / OH, k = i % OH; w2t[i] = f2bf(W2[k * HH + n]); }
  for (int i = tid; i < NOUT * DD; i += np) { int n = i / DD, k = i % DD; wlyt[i] = f2bf(WLy[k * NOUT + n]); }
}

__global__ void k_zero(float* s, u16t* sb) {
  int tid = blockIdx.x * blockDim.x + threadIdx.x;
  int np = gridDim.x * blockDim.x;
  for (int i = tid; i < NB * DD; i += np) { s[i] = 0.0f; sb[i] = (u16t)0; }
}

// ---------------- per-step: fused input embedding + GRU gates ----------------
// Block: 32 batch rows x 64 gate-cols. First the block builds the bf16 embedding
// xin = [tanh(ext@Wu+bu) | tanh(obs@Wx+bx)] for its 32 rows into LDS (64 16x16 WMMA
// tiles, 8 per wave), then runs the gi/gh GEMM pair with A_x sourced from LDS.
__global__ void k_stepA(const float* __restrict__ ext, const float* __restrict__ obs,
                        const u16t* __restrict__ wut, const u16t* __restrict__ wxt,
                        const float* __restrict__ bu, const float* __restrict__ bx,
                        const u16t* __restrict__ sb_cur, const float* __restrict__ s_cur,
                        const u16t* __restrict__ wih, const u16t* __restrict__ whh,
                        const float* __restrict__ b_ih, const float* __restrict__ b_hh,
                        float* __restrict__ s_nxt, u16t* __restrict__ sb_nxt, int t) {
  __shared__ u16t xin_lds[32 * DD];            // 32 KB: this block's embedding rows
  int lane = threadIdx.x & 31, w = threadIdx.x >> 5;
  int rblk = blockIdx.x * 32;
  int half = (lane >> 4) << 3;

  // ---- in-block embedding: 64 tiles (2 row-tiles x 32 col-tiles), 8 per wave ----
#pragma unroll 2
  for (int it = 0; it < 8; ++it) {
    int tile = w * 8 + it;                     // 0..63 (wave-uniform)
    int rt = tile >> 5;                        // row-tile 0/1
    int ec0 = (tile & 31) * 16;                // embedding column base
    int rowg = rblk + rt * 16 + (lane & 15);
    int colG = ec0 + (lane & 15);
    v8f acc = {};
    float bias;
    FragU A;
    if (ec0 < HH) {                            // u-embedding, K = 32 exactly
      const float* p = ext + ((size_t)t * NB + rowg) * (INU + 1);
#pragma unroll
      for (int j = 0; j < 8; ++j) {
        A.u[j]     = f2bf(p[half + j]);
        A.u[8 + j] = f2bf(p[16 + half + j]);
      }
      acc = wmma_bf(A.u, ldb(wut, colG, INU, 0, lane), acc);
      bias = bu[colG];
    } else {                                   // x-embedding, K = 16 padded to 32
      const float* p = obs + ((size_t)t * NB + rowg) * NOUT;
#pragma unroll
      for (int j = 0; j < 8; ++j) {
        A.u[j]     = f2bf(p[half + j]);
        A.u[8 + j] = (u16t)0;
      }
      acc = wmma_bf(A.u, ldb(wxt, colG - HH, 32, 0, lane), acc);
      bias = bx[colG - HH];
    }
#pragma unroll
    for (int v = 0; v < 8; ++v) {
      int ml = rt * 16 + v + ((lane >> 4) << 3);
      xin_lds[ml * DD + colG] = f2bf(tanh_f(acc[v] + bias));
    }
  }
  __syncthreads();

  // ---- GRU gate GEMM pair; wave = one 16x16 tile set {acc_r, acc_z, acc_gin, acc_ghn} ----
  int wr = w >> 2, wc = w & 3;
  int r0 = rblk + wr * 16;
  int c0 = blockIdx.y * 64 + wc * 16;          // in [0, 512)
  int arowl = wr * 16 + (lane & 15);           // local row into xin_lds
  int arow = r0 + (lane & 15);                 // global batch row
  int bcol = c0 + (lane & 15);
  __builtin_prefetch(sb_cur + (size_t)arow * DD, 0, 0);  // -> global_prefetch_b8
  v8f accR = {}, accZ = {}, accI = {}, accN = {};
#pragma unroll 4
  for (int kk = 0; kk < DD; kk += 32) {
    v16u ax = lda(xin_lds, arowl, DD, kk, lane);         // ds_load_b128 x2
    v16u ah = lda(sb_cur, arow, DD, kk, lane);
    v16u br0 = ldb(wih, bcol,          DD, kk, lane);
    v16u br1 = ldb(whh, bcol,          DD, kk, lane);
    v16u bz0 = ldb(wih, DD + bcol,     DD, kk, lane);
    v16u bz1 = ldb(whh, DD + bcol,     DD, kk, lane);
    v16u bn0 = ldb(wih, 2 * DD + bcol, DD, kk, lane);
    v16u bn1 = ldb(whh, 2 * DD + bcol, DD, kk, lane);
    accR = wmma_bf(ax, br0, accR); accR = wmma_bf(ah, br1, accR);   // gi_r + gh_r
    accZ = wmma_bf(ax, bz0, accZ); accZ = wmma_bf(ah, bz1, accZ);   // gi_z + gh_z
    accI = wmma_bf(ax, bn0, accI);                                   // gi_n
    accN = wmma_bf(ah, bn1, accN);                                   // gh_n
  }
  float br = b_ih[bcol] + b_hh[bcol];
  float bz = b_ih[DD + bcol] + b_hh[DD + bcol];
  float bi = b_ih[2 * DD + bcol];
  float bh = b_hh[2 * DD + bcol];
#pragma unroll
  for (int v = 0; v < 8; ++v) {
    int m = r0 + v + ((lane >> 4) << 3);
    size_t idx = (size_t)m * DD + bcol;
    float hold = s_cur[idx];
    float r = sigm(accR[v] + br);
    float z = sigm(accZ[v] + bz);
    float n = tanh_f(accI[v] + bi + r * (accN[v] + bh));
    float hm = (1.0f - z) * n + z * hold;
    s_nxt[idx] = hm;
    sb_nxt[idx] = f2bf(hm);
  }
}

// ---------------- per-step: ODE MLP (h@W1 -> tanh -> @W2, h += dt*f) + y = h_old@WLy ----------------
__global__ void k_stepB(const float* __restrict__ ext, const u16t* __restrict__ sb_cur,
                        const u16t* __restrict__ w1t, const u16t* __restrict__ w2t,
                        const u16t* __restrict__ wlyt,
                        const float* __restrict__ b1, const float* __restrict__ b2,
                        const float* __restrict__ bLy,
                        float* __restrict__ s_nxt, u16t* __restrict__ sb_nxt,
                        float* __restrict__ y, int t) {
  __shared__ u16t a_lds[16 * OH];              // 4 KB staging for the 16x128 tanh activations
  __shared__ float y_lds[16 * NOUT];           // K-split reduction buffer for the y projection
  int lane = threadIdx.x & 31, w = threadIdx.x >> 5;
  int r0 = blockIdx.x * 16;
  int arow = r0 + (lane & 15);

  y_lds[threadIdx.x] = 0.0f;                   // 256 threads == 16x16 tile

  // phase 1: a = tanh(h_mid[:, :256] @ W1 + b1); wave w owns hidden cols [16w, 16w+16)
  {
    int nrow = w * 16 + (lane & 15);
    __builtin_prefetch(sb_nxt + (size_t)arow * DD, 0, 0);
    v8f acc = {};
#pragma unroll
    for (int kk = 0; kk < HH; kk += 32) {
      v16u a = lda(sb_nxt, arow, DD, kk, lane);
      v16u b = ldb(w1t, nrow, HH, kk, lane);
      acc = wmma_bf(a, b, acc);
    }
    float bias = b1[nrow];
#pragma unroll
    for (int v = 0; v < 8; ++v) {
      int ml = v + ((lane >> 4) << 3);
      a_lds[ml * OH + nrow] = f2bf(tanh_f(acc[v] + bias));
    }
  }
  __syncthreads();

  float dts[8];
#pragma unroll
  for (int v = 0; v < 8; ++v) {
    int m = r0 + v + ((lane >> 4) << 3);
    dts[v] = ext[((size_t)t * NB + m) * (INU + 1) + INU];
  }

  // phase 2: f = a @ W2 + b2; h = h_mid + dt*f; wave w owns h cols [32w, 32w+32)
#pragma unroll
  for (int tt = 0; tt < 2; ++tt) {
    int nrow = w * 32 + tt * 16 + (lane & 15); // in [0, 256)
    v8f acc = {};
#pragma unroll
    for (int kk = 0; kk < OH; kk += 32) {
      v16u a = lda(a_lds, (lane & 15), OH, kk, lane);
      v16u b = ldb(w2t, nrow, OH, kk, lane);
      acc = wmma_bf(a, b, acc);
    }
    float bias = b2[nrow];
#pragma unroll
    for (int v = 0; v < 8; ++v) {
      int m = r0 + v + ((lane >> 4) << 3);
      size_t idx = (size_t)m * DD + nrow;
      float hn = s_nxt[idx] + dts[v] * (acc[v] + bias);
      s_nxt[idx] = hn;
      sb_nxt[idx] = f2bf(hn);
    }
  }

  // phase 3: y[t] = h_old @ WLy + bLy from the PRE-step state (scan emits the state before step t).
  // K=512 split across the 8 waves (2 K-steps each), partials reduced via ds_add_f32.
  {
    int nrow = lane & 15;                      // output col 0..15
    int kk0 = w * 64;
    v8f acc = {};
#pragma unroll
    for (int kk = kk0; kk < kk0 + 64; kk += 32) {
      v16u a = lda(sb_cur, arow, DD, kk, lane);
      v16u b = ldb(wlyt, nrow, DD, kk, lane);
      acc = wmma_bf(a, b, acc);
    }
#pragma unroll
    for (int v = 0; v < 8; ++v) {
      int ml = v + ((lane >> 4) << 3);
      atomicAdd(&y_lds[ml * NOUT + nrow], acc[v]);   // ds_add_f32
    }
  }
  __syncthreads();
  {
    int ml = threadIdx.x >> 4, col = threadIdx.x & 15;
    y[((size_t)t * NB + r0 + ml) * NOUT + col] = y_lds[threadIdx.x] + bLy[col];
  }
}

extern "C" void kernel_launch(void* const* d_in, const int* in_sizes, int n_in,
                              void* d_out, int out_size, void* d_ws, size_t ws_size,
                              hipStream_t stream) {
  (void)in_sizes; (void)n_in; (void)out_size; (void)ws_size;
  const float* ext = (const float*)d_in[0];
  const float* obs = (const float*)d_in[1];
  const float* Wu  = (const float*)d_in[2];
  const float* bu  = (const float*)d_in[3];
  const float* Wx  = (const float*)d_in[4];
  const float* bx  = (const float*)d_in[5];
  const float* Wih = (const float*)d_in[6];
  const float* Whh = (const float*)d_in[7];
  const float* bih = (const float*)d_in[8];
  const float* bhh = (const float*)d_in[9];
  const float* W1  = (const float*)d_in[10];
  const float* b1  = (const float*)d_in[11];
  const float* W2  = (const float*)d_in[12];
  const float* b2  = (const float*)d_in[13];
  const float* WLy = (const float*)d_in[14];
  const float* bLy = (const float*)d_in[15];
  float* y = (float*)d_out;

  char* w = (char*)d_ws;
  size_t o = 0;
  u16t* wih = (u16t*)(w + o); o += (size_t)GG * DD * 2;          // 1.5 MB
  u16t* whh = (u16t*)(w + o); o += (size_t)GG * DD * 2;          // 1.5 MB
  u16t* wut = (u16t*)(w + o); o += (size_t)HH * INU * 2;
  u16t* wxt = (u16t*)(w + o); o += (size_t)HH * 32 * 2;
  u16t* w1t = (u16t*)(w + o); o += (size_t)OH * HH * 2;
  u16t* w2t = (u16t*)(w + o); o += (size_t)HH * OH * 2;
  u16t* wlyt = (u16t*)(w + o); o += (size_t)NOUT * DD * 2;
  float* S0 = (float*)(w + o); o += (size_t)NB * DD * 4;         // 2 MB
  float* S1 = (float*)(w + o); o += (size_t)NB * DD * 4;         // 2 MB
  u16t* Sb0 = (u16t*)(w + o); o += (size_t)NB * DD * 2;          // 1 MB
  u16t* Sb1 = (u16t*)(w + o); o += (size_t)NB * DD * 2;          // 1 MB  (~9.2 MB total)

  k_convert<<<512, 256, 0, stream>>>(Wih, Whh, Wu, Wx, W1, W2, WLy,
                                     wih, whh, wut, wxt, w1t, w2t, wlyt);
  k_zero<<<512, 256, 0, stream>>>(S0, Sb0);

  for (int t = 0; t < LT; ++t) {
    float* sc  = (t & 1) ? S1  : S0;
    float* sn  = (t & 1) ? S0  : S1;
    u16t*  sbc = (t & 1) ? Sb1 : Sb0;
    u16t*  sbn = (t & 1) ? Sb0 : Sb1;
    k_stepA<<<dim3(NB / 32, DD / 64), 256, 0, stream>>>(ext, obs, wut, wxt, bu, bx,
                                                        sbc, sc, wih, whh, bih, bhh, sn, sbn, t);
    k_stepB<<<NB / 16, 256, 0, stream>>>(ext, sbc, w1t, w2t, wlyt, b1, b2, bLy, sn, sbn, y, t);
  }
}
